// GlobalAttention_23476291239984
// MI455X (gfx1250) — matrix operand order repaired and checked by reading, not compiled, not run on hardware
//
#include <hip/hip_runtime.h>
#include <math.h>

// GlobalAttention, MI455X (gfx1250, wave32, WMMA f32_16x16x32_f16 + async LDS copy)
// B=1, R=256, S=1024, D=256 (=H*32), H=8, kd=vd=32, OUT=256

#define R_DIM 256
#define S_DIM 1024
#define D_DIM 256
#define EPSV 9.5367431640625e-07f      // 2^-20
#define KEY_SCALE 0.17677669529663687f // 32^-0.5

#define XPITCH 264   // f16 activation tiles: row stride (264*2 B -> conflict-free b128)
#define WPITCH 40    // transposed-W tiles: 32 k + 8 pad halfwords (80 B rows, 16B aligned)
#define KVPITCH 72   // per-chunk k|v tile pitch

#define WG_TILE  (256 * WPITCH)  // 10240 f16 per k-block tile of Wg/Wo
#define WKV_TILE (64 * WPITCH)   // 2560  f16 per k-block tile of Wk|Wv

typedef __attribute__((ext_vector_type(16))) _Float16 v16h;
typedef __attribute__((ext_vector_type(8)))  _Float16 v8h;
typedef __attribute__((ext_vector_type(8)))  float    v8f;

union V16U { v16h v; v8h h[2]; };

// A fragment (16-bit 16x32, ISA 7.12.2): lane l holds M=l&15;
// elems 0..7 -> K = kbase + 8*(l>=16) + 0..7 ; elems 8..15 -> +16
__device__ __forceinline__ v16h load_a_frag(const _Float16* X, int pitch, int lane,
                                            int m0, int kbase) {
  const _Float16* p = X + (m0 + (lane & 15)) * pitch + kbase + ((lane >> 4) << 3);
  V16U u;
  u.h[0] = *(const v8h*)(p);
  u.h[1] = *(const v8h*)(p + 16);
  return u.v;
}

// B fragment (16-bit 32x16): lane l holds N=l&15; elems 0..7 -> K = 8*(l>=16)+0..7; elems 8..15 -> +16.
// W stored transposed as Wt[n][k]; works from LDS or global (L2-resident tiles).
__device__ __forceinline__ v16h load_b_frag(const _Float16* W, int pitch, int lane, int n0) {
  const _Float16* p = W + (n0 + (lane & 15)) * pitch + ((lane >> 4) << 3);
  V16U u;
  u.h[0] = *(const v8h*)(p);
  u.h[1] = *(const v8h*)(p + 16);
  return u.v;
}

__device__ __forceinline__ v8f wmma16(v16h a, v16h b, v8f c) {
  return __builtin_amdgcn_wmma_f32_16x16x32_f16(false, a, false, b, (short)0, c,
                                                false, false);
}

// Async global->LDS copy of `n128` 16-byte chunks (ASYNCcnt-tracked TDM-lite path).
__device__ __forceinline__ void async_copy_b128(const void* gsrc, void* ldst,
                                                int n128, int tid, int nthreads) {
  unsigned long long gbase = (unsigned long long)gsrc;
  unsigned lbase = (unsigned)(unsigned long long)ldst;  // low 32 bits = LDS offset
  for (int i = tid; i < n128; i += nthreads) {
    unsigned long long ga = gbase + (unsigned long long)i * 16u;
    unsigned la = lbase + (unsigned)i * 16u;
    asm volatile("global_load_async_to_lds_b128 %0, %1, off"
                 :: "v"(la), "v"(ga) : "memory");
  }
  asm volatile("s_wait_asynccnt 0x0" ::: "memory");
}

// ---------------- Kernel P: one-time weight repack fp32 -> f16 fragment layout
// grid (8 kb, 3): y==0 Wg, y==1 Wo, y==2 Wk|Wv.  Pad columns zeroed.
__global__ __launch_bounds__(256) void k_prep(const float* __restrict__ Wg,
                                              const float* __restrict__ Wo,
                                              const float* __restrict__ Wk,
                                              const float* __restrict__ Wv,
                                              _Float16* __restrict__ wgT,
                                              _Float16* __restrict__ woT,
                                              _Float16* __restrict__ wkvT) {
  int kb = blockIdx.x, which = blockIdx.y, tid = threadIdx.x;
  if (which < 2) {
    const float* W = which ? Wo : Wg;
    _Float16* dst = (which ? woT : wgT) + (size_t)kb * WG_TILE;
    for (int i = tid; i < WG_TILE; i += 256) {
      int n = i / WPITCH, kk = i % WPITCH;
      dst[i] = (kk < 32) ? (_Float16)W[(size_t)(kb * 32 + kk) * 256 + n]
                         : (_Float16)0.f;
    }
  } else {
    _Float16* dst = wkvT + (size_t)kb * WKV_TILE;
    for (int i = tid; i < WKV_TILE; i += 256) {
      int n = i / WPITCH, kk = i % WPITCH;
      float v = 0.f;
      if (kk < 32)
        v = (n < 32) ? Wk[(size_t)(kb * 32 + kk) * 32 + n]
                     : Wv[(size_t)(kb * 32 + kk) * 32 + (n - 32)];
      dst[i] = (_Float16)v;
    }
  }
}

// ---------------- Kernel A1: masked pool partial sums (deterministic, no atomics)
__global__ __launch_bounds__(256) void k_pool_partial(const float* __restrict__ q_data,
                                                      const float* __restrict__ q_mask,
                                                      float* __restrict__ part) {
  int r = blockIdx.y, cx = blockIdx.x, d = threadIdx.x;
  int s0 = cx * 128;
  const float* qp = q_data + ((size_t)r * S_DIM + s0) * D_DIM + d;
  const float* mp = q_mask + (size_t)r * S_DIM + s0;
  float acc = 0.f, msum = 0.f;
  for (int s = 0; s < 128; ++s) {
    float mk = mp[s];
    acc += qp[(size_t)s * D_DIM] * mk;
    msum += mk;
  }
  float* dst = part + ((size_t)r * 8 + cx) * 257;
  dst[d] = acc;
  if (d == 0) dst[256] = msum;
}

// ---------------- Kernel A2: finish pool, q = (q_avg @ Wq) * key_scale
__global__ __launch_bounds__(256) void k_pool_project(const float* __restrict__ part,
                                                      const float* __restrict__ Wq,
                                                      float* __restrict__ qout) {
  __shared__ float qa[256];
  int r = blockIdx.x, d = threadIdx.x;
  float tot = 0.f, ms = 0.f;
  for (int c = 0; c < 8; ++c) {
    const float* src = part + ((size_t)r * 8 + c) * 257;
    tot += src[d];
    ms  += src[256];
  }
  qa[d] = tot / (ms + EPSV);
  __syncthreads();
  float acc = 0.f;
  for (int kk = 0; kk < 256; ++kk)
    acc += qa[kk] * Wq[kk * 256 + d];
  qout[(size_t)r * 256 + d] = acc * KEY_SCALE;
}

// ---------------- Kernel B: per-r K/V projection (WMMA) + flash-attention softmax
// One block per r; 32-row s-chunks (1 s per lane). WkvT staged once via async copy.
// Wave h owns head h in the softmax phase; GEMM tile: mt = w&1, nt = w>>1.
__global__ __launch_bounds__(256) void k_attention(const float* __restrict__ m_data,
                                                   const float* __restrict__ bias,
                                                   const _Float16* __restrict__ wkvT,
                                                   const float* __restrict__ qvec,
                                                   float* __restrict__ wa) {
  __shared__ __align__(16) _Float16 Mh[32 * XPITCH];        // 16896 B
  __shared__ __align__(16) _Float16 WkvS[8 * WKV_TILE];     // 40960 B (all 8 k-tiles)
  __shared__ __align__(16) _Float16 kvc[32 * KVPITCH];      //  4608 B
  __shared__ float qsh[256];                                //  1024 B
  __shared__ float pch[8 * 32];                             //  1024 B

  int r = blockIdx.x;
  int tid = threadIdx.x;
  int w = tid >> 5, lane = tid & 31;
  int h = w;
  int mt = w & 1, nt = w >> 1;

  qsh[tid] = qvec[(size_t)r * 256 + tid];
  // one-shot async stage of all Wkv^T k-tiles (2560 x 16B)
  async_copy_b128(wkvT, WkvS, 8 * WKV_TILE / 8, tid, 256);
  __syncthreads();

  float mrun = -1.0e30f, srun = 0.f, oacc = 0.f;   // lane holds value-dim d = lane

  for (int c = 0; c < 32; ++c) {
    int s0 = c * 32;
    __syncthreads();
    // stage m_data chunk (32 x 256) into LDS as f16
    for (int i4 = tid; i4 < 32 * 64; i4 += 256) {
      int row = i4 >> 6, c4 = (i4 & 63) << 2;
      float4 f = *(const float4*)(m_data + ((size_t)r * S_DIM + s0 + row) * D_DIM + c4);
      _Float16* dst = &Mh[row * XPITCH + c4];
      dst[0] = (_Float16)f.x; dst[1] = (_Float16)f.y;
      dst[2] = (_Float16)f.z; dst[3] = (_Float16)f.w;
    }
    __syncthreads();
    v8f acc = {};
    #pragma unroll
    for (int kb = 0; kb < 8; ++kb) {
      v16h a = load_a_frag(Mh, XPITCH, lane, mt * 16, kb * 32);
      v16h b = load_b_frag(WkvS + kb * WKV_TILE, WPITCH, lane, nt * 16);
      acc = wmma16(a, b, acc);
    }
    __syncthreads();
    // write k|v chunk (C layout: VGPR i -> row mt*16 + i + 8*(lane>=16))
    {
      int rb = mt * 16 + ((lane >> 4) << 3);
      int c0 = nt * 16 + (lane & 15);
      #pragma unroll
      for (int i = 0; i < 8; ++i)
        kvc[(rb + i) * KVPITCH + c0] = (_Float16)acc[i];
    }
    __syncthreads();
    // logits for head h: s = s0 + lane
    float lg = bias[(size_t)r * S_DIM + s0 + lane];
    #pragma unroll 8
    for (int d = 0; d < 32; ++d)
      lg += qsh[h * 32 + d] * (float)kvc[lane * KVPITCH + d];
    // online softmax update (wave-level reductions)
    float mc = lg;
    for (int off = 16; off > 0; off >>= 1) mc = fmaxf(mc, __shfl_xor(mc, off, 32));
    float mnew  = fmaxf(mrun, mc);
    float scale = __expf(mrun - mnew);
    float p = __expf(lg - mnew);
    float ps = p;
    for (int off = 16; off > 0; off >>= 1) ps += __shfl_xor(ps, off, 32);
    srun = srun * scale + ps;
    pch[h * 32 + lane] = p;
    __syncthreads();
    float vacc = 0.f;
    for (int s = 0; s < 32; ++s)
      vacc += pch[h * 32 + s] * (float)kvc[s * KVPITCH + 32 + lane];
    oacc = oacc * scale + vacc;
    mrun = mnew;
  }
  wa[(size_t)r * 256 + h * 32 + lane] = oacc / srun;
}

// ---------------- Kernel C: fused  sigmoid(X@Wg+bg)*wa  ->  @Wo+bo
// grid (S/64, R). B fragments come straight from the pre-packed f16 tiles in L2,
// so the k-loops run barrier-free; only the Xh tile lives in LDS (34 KB).
__global__ __launch_bounds__(256) void k_gate_out(const float* __restrict__ q_data,
                                                  const _Float16* __restrict__ wgT,
                                                  const float* __restrict__ bg,
                                                  const _Float16* __restrict__ woT,
                                                  const float* __restrict__ bo,
                                                  const float* __restrict__ wa,
                                                  float* __restrict__ out) {
  __shared__ __align__(16) _Float16 Xh[64 * XPITCH];    // X tile, then gated tile
  int r = blockIdx.y, sc = blockIdx.x;
  int s0 = sc * 64;
  int tid = threadIdx.x, w = tid >> 5, lane = tid & 31;
  int nt0 = w << 1;   // wave w owns N-tiles {2w, 2w+1}, all 4 M-tiles

  for (int i4 = tid; i4 < 4096; i4 += 256) {
    int row = i4 >> 6, c4 = (i4 & 63) << 2;
    float4 f = *(const float4*)(q_data + ((size_t)r * S_DIM + s0 + row) * D_DIM + c4);
    _Float16* dst = &Xh[row * XPITCH + c4];
    dst[0] = (_Float16)f.x; dst[1] = (_Float16)f.y;
    dst[2] = (_Float16)f.z; dst[3] = (_Float16)f.w;
  }
  __syncthreads();

  v8f vzero = {};
  v8f acc[4][2];
  #pragma unroll
  for (int m = 0; m < 4; ++m) { acc[m][0] = vzero; acc[m][1] = vzero; }

  // ---- stage 1: X @ Wg (B frags from global/L2, barrier-free)
  #pragma unroll 2
  for (int kb = 0; kb < 8; ++kb) {
    const _Float16* wt = wgT + (size_t)kb * WG_TILE;
    v16h b0 = load_b_frag(wt, WPITCH, lane, nt0 * 16);
    v16h b1 = load_b_frag(wt, WPITCH, lane, (nt0 + 1) * 16);
    #pragma unroll
    for (int m = 0; m < 4; ++m) {
      v16h a = load_a_frag(Xh, XPITCH, lane, m * 16, kb * 32);
      acc[m][0] = wmma16(a, b0, acc[m][0]);
      acc[m][1] = wmma16(a, b1, acc[m][1]);
    }
  }
  __syncthreads();
  // ---- epilogue 1: gated = sigmoid(acc + bg) * wa[r]  -> back into Xh as f16
  #pragma unroll
  for (int j = 0; j < 2; ++j) {
    int cN = (nt0 + j) * 16 + (lane & 15);
    float bgv = bg[cN];
    float wav = wa[(size_t)r * 256 + cN];
    #pragma unroll
    for (int m = 0; m < 4; ++m) {
      int rb = m * 16 + ((lane >> 4) << 3);
      #pragma unroll
      for (int i = 0; i < 8; ++i) {
        float x = acc[m][j][i] + bgv;
        float g = 1.f / (1.f + __expf(-x));
        Xh[(rb + i) * XPITCH + cN] = (_Float16)(g * wav);
      }
    }
  }
  __syncthreads();
  #pragma unroll
  for (int m = 0; m < 4; ++m) { acc[m][0] = vzero; acc[m][1] = vzero; }

  // ---- stage 2: gated @ Wo
  #pragma unroll 2
  for (int kb = 0; kb < 8; ++kb) {
    const _Float16* wt = woT + (size_t)kb * WG_TILE;
    v16h b0 = load_b_frag(wt, WPITCH, lane, nt0 * 16);
    v16h b1 = load_b_frag(wt, WPITCH, lane, (nt0 + 1) * 16);
    #pragma unroll
    for (int m = 0; m < 4; ++m) {
      v16h a = load_a_frag(Xh, XPITCH, lane, m * 16, kb * 32);
      acc[m][0] = wmma16(a, b0, acc[m][0]);
      acc[m][1] = wmma16(a, b1, acc[m][1]);
    }
  }
  // ---- epilogue 2: + bo, store fp32
  #pragma unroll
  for (int j = 0; j < 2; ++j) {
    int cN = (nt0 + j) * 16 + (lane & 15);
    float bov = bo[cN];
    #pragma unroll
    for (int m = 0; m < 4; ++m) {
      int rb = m * 16 + ((lane >> 4) << 3);
      #pragma unroll
      for (int i = 0; i < 8; ++i)
        out[((size_t)r * S_DIM + s0 + rb + i) * D_DIM + cN] = acc[m][j][i] + bov;
    }
  }
}

extern "C" void kernel_launch(void* const* d_in, const int* in_sizes, int n_in,
                              void* d_out, int out_size, void* d_ws, size_t ws_size,
                              hipStream_t stream) {
  (void)in_sizes; (void)n_in; (void)out_size; (void)ws_size;
  const float* q_data = (const float*)d_in[0];
  const float* m_data = (const float*)d_in[1];
  const float* q_mask = (const float*)d_in[2];
  const float* bias   = (const float*)d_in[3];
  const float* Wq     = (const float*)d_in[4];
  const float* Wk     = (const float*)d_in[5];
  const float* Wv     = (const float*)d_in[6];
  const float* Wo     = (const float*)d_in[7];
  const float* bo     = (const float*)d_in[8];
  const float* Wg     = (const float*)d_in[9];
  const float* bg     = (const float*)d_in[10];
  float* out  = (float*)d_out;

  // workspace layout: fp32 partials | q | weighted_avg, then 16B-aligned f16 tiles
  float* part = (float*)d_ws;
  float* qv   = part + (size_t)8 * R_DIM * 257;
  float* wav  = qv + (size_t)R_DIM * D_DIM;
  _Float16* wgT  = (_Float16*)(wav + (size_t)R_DIM * D_DIM);   // 8 * 10240 f16
  _Float16* woT  = wgT + (size_t)8 * WG_TILE;                  // 8 * 10240 f16
  _Float16* wkvT = woT + (size_t)8 * WG_TILE;                  // 8 * 2560  f16
  // total ws: ~2.6 MB fp32 + ~370 KB f16

  k_prep<<<dim3(8, 3), 256, 0, stream>>>(Wg, Wo, Wk, Wv, wgT, woT, wkvT);
  k_pool_partial<<<dim3(8, R_DIM), 256, 0, stream>>>(q_data, q_mask, part);
  k_pool_project<<<dim3(R_DIM), 256, 0, stream>>>(part, Wq, qv);
  k_attention<<<dim3(R_DIM), 256, 0, stream>>>(m_data, bias, wkvT, qv, wav);
  k_gate_out<<<dim3(S_DIM / 64, R_DIM), 256, 0, stream>>>(q_data, wgT, bg, woT, bo, wav, out);
}
